// MetabolismProcessor_76106820485633
// MI455X (gfx1250) — compile-verified, matrix operand order checked by reading, not run on hardware
//
#include <hip/hip_runtime.h>

// ---------------------------------------------------------------------------
// Problem constants (match reference)
// ---------------------------------------------------------------------------
static constexpr int CDIM    = 128;
static constexpr int NH      = 2;     // heads
static constexpr int HD      = 64;    // head dim
static constexpr int MAXSET  = 48;
static constexpr int CAP     = 64;    // slot capacity per group (>= MAXSET)
static constexpr int N_GENE  = 12000;
static constexpr int N_RXN   = 10000;
static constexpr int N_MET   = 6000;
static constexpr int E_GPR   = 80000;
static constexpr int E_MET   = 120000;
static constexpr int WMAT    = CDIM * CDIM;   // 16384 elements per weight matrix

// ---------------------------------------------------------------------------
// WMMA plumbing (gfx1250, wave32). bf16 operands, f32 accumulate.
// ---------------------------------------------------------------------------
typedef __bf16 bf16;
typedef __attribute__((ext_vector_type(16))) __bf16 v16bf;
typedef __attribute__((ext_vector_type(8)))  float  v8f;

#define DEVI static __device__ __forceinline__

DEVI v8f wmma_bf16(v16bf a, v16bf b, v8f c) {
  // D = A(16x32) * B(32x16) + C, f32 accumulate
  return __builtin_amdgcn_wmma_f32_16x16x32_bf16(false, a, false, b, (short)0, c,
                                                 false, false);
}

DEVI bf16 to_bf16(float v) { return (bf16)v; }
DEVI bf16 to_bf16(bf16 v) { return v; }

// K-index pattern within a lane (ISA 7.12.2 16-bit A layout:
// lanes 0-15 hold K 0..7,16..23; lanes 16-31 hold K 8..15,24..31)
DEVI int kpat(int j, int lane) { return ((j < 8) ? j : (j + 8)) + ((lane >> 4) << 3); }

// A fragment: rows = M (one per lane in each half), cols = K. Zero outside lims.
template <class T>
DEVI v16bf a_frag(const T* s, int ld, int mrow, int kcol, int mlim, int klim) {
  const int lane = threadIdx.x & 31;
  const int m = mrow + (lane & 15);
  v16bf a;
#pragma unroll
  for (int j = 0; j < 16; ++j) {
    const int k = kcol + kpat(j, lane);
    bf16 v = (bf16)0.0f;
    if (m < mlim && k < klim) v = to_bf16(s[(size_t)m * ld + k]);
    a[j] = v;
  }
  return a;
}

// B fragment with B[k][n] = S[n][k]   (i.e. multiply by S^T; S row-major [n][k])
template <class T>
DEVI v16bf bT_frag(const T* s, int ld, int kcol, int ncol, int klim) {
  const int lane = threadIdx.x & 31;
  const int n = ncol + (lane & 15);
  v16bf b;
#pragma unroll
  for (int j = 0; j < 16; ++j) {
    const int k = kcol + kpat(j, lane);
    bf16 v = (bf16)0.0f;
    if (k < klim) v = to_bf16(s[(size_t)n * ld + k]);
    b[j] = v;
  }
  return b;
}

// B fragment with B[k][n] = S[k][n]   (S row-major [k][n])
template <class T>
DEVI v16bf b_frag(const T* s, int ld, int kcol, int ncol, int klim) {
  const int lane = threadIdx.x & 31;
  const int n = ncol + (lane & 15);
  v16bf b;
#pragma unroll
  for (int j = 0; j < 16; ++j) {
    const int k = kcol + kpat(j, lane);
    bf16 v = (bf16)0.0f;
    if (k < klim) v = to_bf16(s[(size_t)k * ld + n]);
    b[j] = v;
  }
  return b;
}

// D layout (ISA 7.12.2): VGPR j -> row mrow + j (+8 for lanes 16-31), col = lane&15
DEVI void store_bf16(bf16* dst, int ld, int mrow, int ncol, v8f acc,
                     const float* bias, int mlim, bool relu) {
  const int lane = threadIdx.x & 31;
  const int col = ncol + (lane & 15);
  const int roff = (lane >> 4) << 3;
  const float bv = bias ? bias[col] : 0.0f;
#pragma unroll
  for (int j = 0; j < 8; ++j) {
    const int r = mrow + roff + j;
    float v = acc[j] + bv;
    if (relu && v < 0.0f) v = 0.0f;
    dst[(size_t)r * ld + col] = (bf16)((r < mlim) ? v : 0.0f);
  }
}

DEVI void store_att(float* att, int ld, int mrow, int ncol, v8f acc, int kval, int mlim) {
  const int lane = threadIdx.x & 31;
  const int col = ncol + (lane & 15);
  const int roff = (lane >> 4) << 3;
#pragma unroll
  for (int j = 0; j < 8; ++j) {
    const int r = mrow + roff + j;
    float v = 0.0f;
    if (r < mlim) v = acc[j] * 0.125f + ((col < kval) ? 0.0f : -1.0e9f);
    att[(size_t)r * ld + col] = v;
  }
}

DEVI void store_add(float* dst, int ld, int mrow, int ncol, v8f acc,
                    const float* bias, int addlim) {
  const int lane = threadIdx.x & 31;
  const int col = ncol + (lane & 15);
  const int roff = (lane >> 4) << 3;
#pragma unroll
  for (int j = 0; j < 8; ++j) {
    const int r = mrow + roff + j;
    if (r < addlim) dst[(size_t)r * ld + col] += acc[j] + bias[col];
  }
}

// Row softmax over att[ROWS][W]
template <int ROWS, int W>
DEVI void softmax_rows(float* att) {
  for (int r = threadIdx.x; r < ROWS; r += 128) {
    float* row = att + (size_t)r * W;
    float mx = -3.0e38f;
    for (int j = 0; j < W; ++j) mx = fmaxf(mx, row[j]);
    float s = 0.0f;
    for (int j = 0; j < W; ++j) { float e = __expf(row[j] - mx); row[j] = e; s += e; }
    const float inv = 1.0f / (s + 1e-16f);
    for (int j = 0; j < W; ++j) row[j] *= inv;
  }
}

// LayerNorm over 128 channels, one wave per row (wave32, 4 values/lane)
template <int ROWS>
DEVI void ln_rows(float* xs, const float* gam, const float* bet) {
  const int lane = threadIdx.x & 31;
  const int wv = threadIdx.x >> 5;
  for (int r = wv; r < ROWS; r += 4) {
    float* row = xs + (size_t)r * CDIM;
    float v0 = row[lane], v1 = row[lane + 32], v2 = row[lane + 64], v3 = row[lane + 96];
    float s = v0 + v1 + v2 + v3;
    float q = v0 * v0 + v1 * v1 + v2 * v2 + v3 * v3;
#pragma unroll
    for (int o = 16; o > 0; o >>= 1) { s += __shfl_xor(s, o); q += __shfl_xor(q, o); }
    const float mean = s * (1.0f / 128.0f);
    const float var = q * (1.0f / 128.0f) - mean * mean;
    const float inv = rsqrtf(var + 1e-5f);
    row[lane]      = (v0 - mean) * inv * gam[lane]      + bet[lane];
    row[lane + 32] = (v1 - mean) * inv * gam[lane + 32] + bet[lane + 32];
    row[lane + 64] = (v2 - mean) * inv * gam[lane + 64] + bet[lane + 64];
    row[lane + 96] = (v3 - mean) * inv * gam[lane + 96] + bet[lane + 96];
  }
}

// ---------------------------------------------------------------------------
// MAB parameters: bf16 weights (pre-converted once per launch), f32 vectors
// ---------------------------------------------------------------------------
struct MabP {
  const bf16  *wq, *wk, *wv, *wo, *lw;
  const float *bq, *bk, *bv, *bo, *lb, *g1, *b1, *g2, *b2;
};

// ---------------------------------------------------------------------------
// Fused MAB: one group per 128-thread block (4 waves). Everything in LDS.
//   SQ/SK: real query/key rows; YMASK: key mask from counts; XMASK: h masked;
//   SEPY: y != x (separate key/value source buffer).
// ---------------------------------------------------------------------------
template <int SQ, int SK, bool YMASK, bool XMASK, bool SEPY>
__global__ __launch_bounds__(128)
void mab_kernel(const float* __restrict__ xin, long long xstride,
                const float* __restrict__ yin, long long ystride,
                const int* __restrict__ counts,
                float* __restrict__ outp, long long ostride,
                MabP p, int B) {
  constexpr int SQP = (SQ + 15) & ~15;
  constexpr int SKP = (SK + 15) & ~15;
  constexpr int KD  = (SKP + 31) & ~31;              // K extent for attn@V
  constexpr int MTQ = SQP / 16;
  constexpr int MTK = SKP / 16;
  constexpr int YSZ = SEPY ? SKP * CDIM : 1;
  constexpr int SCR = (2 * SKP * HD > SQP * CDIM) ? 2 * SKP * HD : SQP * CDIM;

  __shared__ float xs[SQP * CDIM];   // query/residual stream (f32)
  __shared__ float ysb[YSZ];         // key/value source when SEPY
  __shared__ bf16  ob[SQP * CDIM];   // concat attention output (bf16)
  __shared__ bf16  qh[SQP * HD];     // per-head Q
  __shared__ bf16  scr[SCR];         // per-head K|V, reused for FFN output
  __shared__ float att[SQP * SKP];   // logits / weights (f32)

  bf16* kh = scr;
  bf16* vh = scr + SKP * HD;

  const int tid = threadIdx.x;
  const int wv = tid >> 5;
  const int g = blockIdx.x;
  if (g >= B) return;

  // Warm WGP$/L2 for this MAB's bf16 weights (global_prefetch_b8)
  {
    const bf16* wlist[5] = {p.wq, p.wk, p.wv, p.wo, p.lw};
#pragma unroll
    for (int wsel = 0; wsel < 5; ++wsel)
      for (int i = tid * 64; i < WMAT; i += 128 * 64)
        __builtin_prefetch(wlist[wsel] + i, 0, 1);
  }

  const float* xg = xin + (size_t)g * (size_t)xstride;
  for (int i = tid; i < SQP * CDIM; i += 128) {
    const int r = i >> 7;
    xs[i] = (r < SQ) ? xg[(size_t)r * CDIM + (i & 127)] : 0.0f;
  }
  if (SEPY) {
    const float* yg = yin + (size_t)g * (size_t)ystride;
    for (int i = tid; i < SKP * CDIM; i += 128) {
      const int r = i >> 7;
      ysb[i] = (r < SK) ? yg[(size_t)r * CDIM + (i & 127)] : 0.0f;
    }
  }
  int mvalid = SK;
  if (YMASK) { const int c = counts[g]; mvalid = (c < SK) ? c : SK; }
  __syncthreads();

  const float* ysrc = SEPY ? ysb : xs;

  for (int h = 0; h < NH; ++h) {
    // ---- Q_h = xs @ Wq_h^T + bq_h ; K_h,V_h from ysrc --------------------
    for (int t = wv; t < MTQ * (HD / 16); t += 4) {
      const int mt = t / (HD / 16), nt = t % (HD / 16);
      v8f acc = {};
#pragma unroll
      for (int kt = 0; kt < 4; ++kt)
        acc = wmma_bf16(a_frag(xs, CDIM, mt * 16, kt * 32, SQP, CDIM),
                        bT_frag(p.wq, CDIM, kt * 32, h * HD + nt * 16, CDIM), acc);
      store_bf16(qh, HD, mt * 16, nt * 16, acc, p.bq + h * HD, SQ, false);
    }
    for (int t = wv; t < MTK * (HD / 16); t += 4) {
      const int mt = t / (HD / 16), nt = t % (HD / 16);
      v8f acc = {};
#pragma unroll
      for (int kt = 0; kt < 4; ++kt)
        acc = wmma_bf16(a_frag(ysrc, CDIM, mt * 16, kt * 32, SKP, CDIM),
                        bT_frag(p.wk, CDIM, kt * 32, h * HD + nt * 16, CDIM), acc);
      store_bf16(kh, HD, mt * 16, nt * 16, acc, p.bk + h * HD, SK, false);
    }
    for (int t = wv; t < MTK * (HD / 16); t += 4) {
      const int mt = t / (HD / 16), nt = t % (HD / 16);
      v8f acc = {};
#pragma unroll
      for (int kt = 0; kt < 4; ++kt)
        acc = wmma_bf16(a_frag(ysrc, CDIM, mt * 16, kt * 32, SKP, CDIM),
                        bT_frag(p.wv, CDIM, kt * 32, h * HD + nt * 16, CDIM), acc);
      store_bf16(vh, HD, mt * 16, nt * 16, acc, p.bv + h * HD, SK, false);
    }
    __syncthreads();

    // ---- logits = Q_h K_h^T / 8 (+mask) ----------------------------------
    for (int t = wv; t < MTQ * MTK; t += 4) {
      const int mt = t / MTK, nt = t % MTK;
      v8f acc = {};
#pragma unroll
      for (int kt = 0; kt < HD / 32; ++kt)
        acc = wmma_bf16(a_frag(qh, HD, mt * 16, kt * 32, SQP, HD),
                        bT_frag(kh, HD, kt * 32, nt * 16, HD), acc);
      store_att(att, SKP, mt * 16, nt * 16, acc, mvalid, SQ);
    }
    __syncthreads();
    softmax_rows<SQ, SKP>(att);
    __syncthreads();

    // ---- O_h = softmax @ V_h --------------------------------------------
    for (int t = wv; t < MTQ * (HD / 16); t += 4) {
      const int mt = t / (HD / 16), nt = t % (HD / 16);
      v8f acc = {};
#pragma unroll
      for (int kt = 0; kt < KD / 32; ++kt)
        acc = wmma_bf16(a_frag(att, SKP, mt * 16, kt * 32, SQP, SKP),
                        b_frag(vh, HD, kt * 32, nt * 16, SKP), acc);
      store_bf16(ob, CDIM, mt * 16, h * HD + nt * 16, acc, (const float*)nullptr,
                 SQP, false);
    }
    __syncthreads();
  }

  // ---- residual: xs += (ob @ Wo^T + bo)  (masked rows skipped) ------------
  const int addlim = XMASK ? ((mvalid < SQ) ? mvalid : SQ) : SQ;
  for (int t = wv; t < MTQ * 8; t += 4) {
    const int mt = t >> 3, nt = t & 7;
    v8f acc = {};
#pragma unroll
    for (int kt = 0; kt < 4; ++kt)
      acc = wmma_bf16(a_frag(ob, CDIM, mt * 16, kt * 32, SQP, CDIM),
                      bT_frag(p.wo, CDIM, kt * 32, nt * 16, CDIM), acc);
    store_add(xs, CDIM, mt * 16, nt * 16, acc, p.bo, addlim);
  }
  __syncthreads();
  ln_rows<SQ>(xs, p.g1, p.b1);
  __syncthreads();

  // ---- FFN: xs += relu(xs @ lw^T + lb) ------------------------------------
  bf16* ffn = scr;
  for (int t = wv; t < MTQ * 8; t += 4) {
    const int mt = t >> 3, nt = t & 7;
    v8f acc = {};
#pragma unroll
    for (int kt = 0; kt < 4; ++kt)
      acc = wmma_bf16(a_frag(xs, CDIM, mt * 16, kt * 32, SQP, CDIM),
                      bT_frag(p.lw, CDIM, kt * 32, nt * 16, CDIM), acc);
    store_bf16(ffn, CDIM, mt * 16, nt * 16, acc, p.lb, SQ, true);
  }
  __syncthreads();
  for (int i = tid; i < SQ * CDIM; i += 128) xs[i] += (float)ffn[i];
  __syncthreads();
  ln_rows<SQ>(xs, p.g2, p.b2);
  __syncthreads();

  float* og = outp + (size_t)g * (size_t)ostride;
  for (int i = tid; i < SQ * CDIM; i += 128) og[i] = xs[i];
}

// ---------------------------------------------------------------------------
// PMA pre-linear: dense <- relu(dense @ W^T + b), per-group in LDS
// ---------------------------------------------------------------------------
__global__ __launch_bounds__(128)
void pma_prep_kernel(float* __restrict__ dense, const bf16* __restrict__ W,
                     const float* __restrict__ bias, int B) {
  __shared__ float xs[MAXSET * CDIM];
  if ((int)blockIdx.x >= B) return;
  float* gptr = dense + (size_t)blockIdx.x * MAXSET * CDIM;
  for (int i = threadIdx.x; i < MAXSET * CDIM; i += 128) xs[i] = gptr[i];
  __syncthreads();
  const int wv = threadIdx.x >> 5;
  for (int t = wv; t < 24; t += 4) {  // 3 x 8 tiles
    const int mt = t >> 3, nt = t & 7;
    v8f acc = {};
#pragma unroll
    for (int kt = 0; kt < 4; ++kt)
      acc = wmma_bf16(a_frag(xs, CDIM, mt * 16, kt * 32, MAXSET, CDIM),
                      bT_frag(W, CDIM, kt * 32, nt * 16, CDIM), acc);
    const int lane = threadIdx.x & 31;
    const int col = nt * 16 + (lane & 15);
    const int roff = (lane >> 4) << 3;
#pragma unroll
    for (int j = 0; j < 8; ++j) {
      const int r = mt * 16 + roff + j;
      float v = acc[j] + bias[col];
      gptr[(size_t)r * CDIM + col] = v > 0.0f ? v : 0.0f;
    }
  }
}

// ---------------------------------------------------------------------------
// Generic (M,128) @ W(128,128)^T GEMM, 64 rows per block
// ---------------------------------------------------------------------------
__global__ __launch_bounds__(128)
void gemm_rows_kernel(const float* __restrict__ in, const bf16* __restrict__ W,
                      float* __restrict__ outp, int M) {
  __shared__ bf16 ab[64 * CDIM];
  const int r0 = blockIdx.x * 64;
  for (int i = threadIdx.x; i < 64 * CDIM; i += 128) {
    const int r = r0 + (i >> 7);
    ab[i] = (bf16)((r < M) ? in[(size_t)r * CDIM + (i & 127)] : 0.0f);
  }
  __syncthreads();
  const int wv = threadIdx.x >> 5;
  for (int t = wv; t < 32; t += 4) {
    const int mt = t >> 3, nt = t & 7;
    v8f acc = {};
#pragma unroll
    for (int kt = 0; kt < 4; ++kt)
      acc = wmma_bf16(a_frag(ab, CDIM, mt * 16, kt * 32, 64, CDIM),
                      bT_frag(W, CDIM, kt * 32, nt * 16, CDIM), acc);
    const int lane = threadIdx.x & 31;
    const int col = nt * 16 + (lane & 15);
    const int roff = (lane >> 4) << 3;
#pragma unroll
    for (int j = 0; j < 8; ++j) {
      const int r = r0 + mt * 16 + roff + j;
      if (r < M) outp[(size_t)r * CDIM + col] = acc[j];
    }
  }
}

// ---------------------------------------------------------------------------
// Once-per-launch f32 -> bf16 weight conversion
// ---------------------------------------------------------------------------
__global__ void convert_bf16_kernel(const float* __restrict__ src,
                                    bf16* __restrict__ dst, int n) {
  const int i = blockIdx.x * blockDim.x + threadIdx.x;
  if (i < n) dst[i] = (bf16)src[i];
}

// ---------------------------------------------------------------------------
// Dense-group construction (stable segment order = ascending edge ids)
// ---------------------------------------------------------------------------
__global__ void fill_slots_kernel(const int* __restrict__ grp, int E,
                                  int* __restrict__ counts, int* __restrict__ slots) {
  const int e = blockIdx.x * blockDim.x + threadIdx.x;
  if (e >= E) return;
  const int g = grp[e];
  const int s = atomicAdd(&counts[g], 1);
  if (s < CAP) slots[(size_t)g * CAP + s] = e;
}

__global__ void sort_slots_kernel(const int* __restrict__ counts,
                                  int* __restrict__ slots, int B) {
  const int g = blockIdx.x * blockDim.x + threadIdx.x;
  if (g >= B) return;
  int n = counts[g]; if (n > CAP) n = CAP;
  int* s = slots + (size_t)g * CAP;
  for (int i = 1; i < n; ++i) {
    const int key = s[i];
    int j = i - 1;
    while (j >= 0 && s[j] > key) { s[j + 1] = s[j]; --j; }
    s[j + 1] = key;
  }
}

__global__ __launch_bounds__(128)
void gather_dense_kernel(const float* __restrict__ feats, const int* __restrict__ e2f,
                         const int* __restrict__ slots, const int* __restrict__ counts,
                         float* __restrict__ dense, int B) {
  const int g = blockIdx.x;
  if (g >= B) return;
  const int c = threadIdx.x;
  int m = counts[g]; if (m > MAXSET) m = MAXSET;
  float* dg = dense + (size_t)g * MAXSET * CDIM;
  for (int pp = 0; pp < MAXSET; ++pp) {
    float v = 0.0f;
    if (pp < m) {
      const int e = slots[(size_t)g * CAP + pp];
      v = feats[(size_t)e2f[e] * CDIM + c];
    }
    dg[(size_t)pp * CDIM + c] = v;
  }
}

__global__ void maskzero_kernel(float* __restrict__ v, const int* __restrict__ counts,
                                int B) {
  const int i = blockIdx.x * blockDim.x + threadIdx.x;
  if (i >= B * CDIM) return;
  if (counts[i >> 7] == 0) v[i] = 0.0f;
}

// ---------------------------------------------------------------------------
// emb row clamp-norm
// ---------------------------------------------------------------------------
__global__ __launch_bounds__(32)
void clampnorm_kernel(const float* __restrict__ emb, float* __restrict__ zm, int N) {
  const int r = blockIdx.x;
  if (r >= N) return;
  const int lane = threadIdx.x;
  const float* row = emb + (size_t)r * CDIM;
  float v0 = row[lane], v1 = row[lane + 32], v2 = row[lane + 64], v3 = row[lane + 96];
  float q = v0 * v0 + v1 * v1 + v2 * v2 + v3 * v3;
#pragma unroll
  for (int o = 16; o > 0; o >>= 1) q += __shfl_xor(q, o);
  const float n = sqrtf(q);
  const float sc = (n > 1.0f) ? 1.0f / (n + 1e-7f) : 1.0f;
  float* orow = zm + (size_t)r * CDIM;
  orow[lane] = v0 * sc; orow[lane + 32] = v1 * sc;
  orow[lane + 64] = v2 * sc; orow[lane + 96] = v3 * sc;
}

// ---------------------------------------------------------------------------
// Stoich conv helpers (segment softmax via monotonic-uint atomicMax)
// ---------------------------------------------------------------------------
DEVI unsigned fenc(float f) {
  const unsigned u = __float_as_uint(f);
  return (u & 0x80000000u) ? ~u : (u | 0x80000000u);
}
DEVI float fdec(unsigned u) {
  const unsigned b = (u & 0x80000000u) ? (u & 0x7FFFFFFFu) : ~u;
  return __uint_as_float(b);
}

__global__ void edge_logits_kernel(const float* __restrict__ xt, const float* __restrict__ et,
                                   const float* __restrict__ attv, const int* __restrict__ src,
                                   const int* __restrict__ dst, float* __restrict__ la,
                                   unsigned* __restrict__ segmax, int E) {
  const int e = blockIdx.x * blockDim.x + threadIdx.x;
  if (e >= E) return;
  const float* xr = xt + (size_t)src[e] * CDIM;
  const float* er = et + (size_t)dst[e] * CDIM;
  float a = 0.0f;
  for (int c = 0; c < CDIM; ++c) a += xr[c] * attv[c];
  for (int c = 0; c < CDIM; ++c) a += er[c] * attv[CDIM + c];
  a = (a >= 0.0f) ? a : 0.2f * a;   // leaky_relu 0.2
  la[e] = a;
  atomicMax(&segmax[dst[e]], fenc(a));
}

__global__ void expsum_kernel(float* __restrict__ la, const unsigned* __restrict__ segmax,
                              float* __restrict__ segsum, const int* __restrict__ dst, int E) {
  const int e = blockIdx.x * blockDim.x + threadIdx.x;
  if (e >= E) return;
  const float m = fdec(segmax[dst[e]]);
  const float ex = __expf(la[e] - m);
  la[e] = ex;
  atomicAdd(&segsum[dst[e]], ex);
}

__global__ void deg_kernel(const float* __restrict__ stoich, const int* __restrict__ src,
                           const int* __restrict__ dst, float* __restrict__ Dn,
                           float* __restrict__ Be, int E) {
  const int e = blockIdx.x * blockDim.x + threadIdx.x;
  if (e >= E) return;
  const float w = fabsf(stoich[e]);
  atomicAdd(&Dn[src[e]], w);
  atomicAdd(&Be[dst[e]], w);
}

__global__ __launch_bounds__(128)
void eout_kernel(const float* __restrict__ xt, const float* __restrict__ la,
                 const float* __restrict__ segsum, const float* __restrict__ Be,
                 const float* __restrict__ stoich, const int* __restrict__ src,
                 const int* __restrict__ dst, float* __restrict__ e_out, int E) {
  const int e = blockIdx.x;
  if (e >= E) return;
  const int c = threadIdx.x;
  const int d = dst[e], s = src[e];
  const float alpha = la[e] / (segsum[d] + 1e-16f);
  const float be = Be[d];
  const float coef = (be > 0.0f ? 1.0f / be : 0.0f) * alpha * stoich[e];
  atomicAdd(&e_out[(size_t)d * CDIM + c], coef * xt[(size_t)s * CDIM + c]);
}

__global__ __launch_bounds__(128)
void nodeout_kernel(const float* __restrict__ e_out, const float* __restrict__ la,
                    const float* __restrict__ segsum, const float* __restrict__ Dn,
                    const float* __restrict__ stoich, const int* __restrict__ src,
                    const int* __restrict__ dst, float* __restrict__ zout, int E) {
  const int e = blockIdx.x;
  if (e >= E) return;
  const int c = threadIdx.x;
  const int d = dst[e], s = src[e];
  const float alpha = la[e] / (segsum[d] + 1e-16f);
  const float dn = Dn[s];
  const float coef = (dn > 0.0f ? 1.0f / dn : 0.0f) * alpha * stoich[e];
  atomicAdd(&zout[(size_t)s * CDIM + c], coef * e_out[(size_t)d * CDIM + c]);
}

__global__ void addbias_kernel(float* __restrict__ z, const float* __restrict__ bias, int N) {
  const int i = blockIdx.x * blockDim.x + threadIdx.x;
  if (i >= N * CDIM) return;
  z[i] += bias[i & 127];
}

// ---------------------------------------------------------------------------
// Host orchestration.
// Input leaf order (insertion-order DFS of setup_inputs() pytree), 131 leaves:
//   0 gene_x, 1 emb,
//   2..60   st_gr  = enc0[14] enc1[14] pma_lin_w pma_lin_b seed pma_mab[14] dec[14]
//   61..119 st_rm  (same layout)
//   120..125 convs = {lin_w, att, bias} x 2
//   126 stoich, 127 gpr_gene_idx, 128 gpr_rxn_idx, 129 met_met_idx, 130 met_rxn_idx
// MAB leaf order: wq wk wv bq bk bv wo bo lw lb ln1_g ln1_b ln2_g ln2_b
// Workspace requirement: ~335 MB.
// ---------------------------------------------------------------------------
extern "C" void kernel_launch(void* const* d_in, const int* in_sizes, int n_in,
                              void* d_out, int out_size, void* d_ws, size_t ws_size,
                              hipStream_t stream) {
  (void)in_sizes; (void)n_in; (void)out_size; (void)ws_size;
  auto F = [&](int i) { return (const float*)d_in[i]; };
  auto I = [&](int i) { return (const int*)d_in[i]; };

  const float* gene_x = F(0);
  const float* emb = F(1);
  const int ST_GR = 2, ST_RM = 61, CONV = 120;
  const float* stoich = F(126);
  const int* gpr_gene = I(127);
  const int* gpr_rxn = I(128);
  const int* met_met = I(129);
  const int* met_rxn = I(130);

  // ---- carve workspace ----
  size_t off = 0;
  auto alloc = [&](size_t bytes) -> void* {
    off = (off + 255) & ~(size_t)255;
    void* p = (char*)d_ws + off;
    off += bytes;
    return p;
  };
  float*    dense  = (float*)alloc((size_t)N_GENE * MAXSET * CDIM * 4);
  int*      counts = (int*)alloc((size_t)N_GENE * 4);
  int*      slots  = (int*)alloc((size_t)N_GENE * CAP * 4);
  float*    H_r    = (float*)alloc((size_t)N_RXN * CDIM * 4);
  float*    Z_r    = (float*)alloc((size_t)N_RXN * CDIM * 4);
  float*    Zm_a   = (float*)alloc((size_t)N_MET * CDIM * 4);
  float*    Zm_b   = (float*)alloc((size_t)N_MET * CDIM * 4);
  float*    xt     = (float*)alloc((size_t)N_MET * CDIM * 4);
  float*    et     = (float*)alloc((size_t)N_RXN * CDIM * 4);
  float*    la     = (float*)alloc((size_t)E_MET * 4);
  unsigned* segmax = (unsigned*)alloc((size_t)N_RXN * 4);
  float*    segsum = (float*)alloc((size_t)N_RXN * 4);
  float*    Dn     = (float*)alloc((size_t)N_MET * 4);
  float*    Be     = (float*)alloc((size_t)N_RXN * 4);
  float*    e_out  = (float*)alloc((size_t)N_RXN * CDIM * 4);
  bf16*     wpool  = (bf16*)alloc((size_t)64 * WMAT * 2);   // up to 64 bf16 matrices

  // ---- convert all 128x128 weight matrices to bf16 once per launch ----
  bf16* wb[131] = {};
  int wcount = 0;
  auto cvt = [&](int leaf) {
    bf16* dst = wpool + (size_t)wcount * WMAT;
    ++wcount;
    convert_bf16_kernel<<<(WMAT + 255) / 256, 256, 0, stream>>>(F(leaf), dst, WMAT);
    wb[leaf] = dst;
  };
  for (int sb = ST_GR; sb <= ST_RM; sb += (ST_RM - ST_GR)) {
    const int mabs[4] = {0, 14, 31, 45};
    for (int mi = 0; mi < 4; ++mi) {
      const int b = sb + mabs[mi];
      cvt(b + 0); cvt(b + 1); cvt(b + 2); cvt(b + 6); cvt(b + 8);  // wq wk wv wo lw
    }
    cvt(sb + 28);  // pma_lin_w
  }
  cvt(CONV + 0);   // conv0 lin_w
  cvt(CONV + 3);   // conv1 lin_w

  auto mk = [&](int b) {
    MabP p;
    p.wq = wb[b + 0]; p.wk = wb[b + 1]; p.wv = wb[b + 2];
    p.wo = wb[b + 6]; p.lw = wb[b + 8];
    p.bq = F(b + 3); p.bk = F(b + 4); p.bv = F(b + 5);
    p.bo = F(b + 7); p.lb = F(b + 9);
    p.g1 = F(b + 10); p.b1 = F(b + 11); p.g2 = F(b + 12); p.b2 = F(b + 13);
    return p;
  };

  auto run_st = [&](const float* feats, const int* grp, const int* e2f, int E, int B,
                    int sb, float* outv) {
    hipMemsetAsync(counts, 0, (size_t)B * 4, stream);
    fill_slots_kernel<<<(E + 255) / 256, 256, 0, stream>>>(grp, E, counts, slots);
    sort_slots_kernel<<<(B + 127) / 128, 128, 0, stream>>>(counts, slots, B);
    gather_dense_kernel<<<B, 128, 0, stream>>>(feats, e2f, slots, counts, dense, B);
    const MabP e0 = mk(sb), e1 = mk(sb + 14), pm = mk(sb + 31), dc = mk(sb + 45);
    const bf16* plw = wb[sb + 28];
    const float* plb = F(sb + 29);
    const float* seed = F(sb + 30);
    mab_kernel<MAXSET, MAXSET, true, true, false><<<B, 128, 0, stream>>>(
        dense, (long long)MAXSET * CDIM, nullptr, 0, counts,
        dense, (long long)MAXSET * CDIM, e0, B);
    mab_kernel<MAXSET, MAXSET, true, true, false><<<B, 128, 0, stream>>>(
        dense, (long long)MAXSET * CDIM, nullptr, 0, counts,
        dense, (long long)MAXSET * CDIM, e1, B);
    pma_prep_kernel<<<B, 128, 0, stream>>>(dense, plw, plb, B);
    mab_kernel<1, MAXSET, true, false, true><<<B, 128, 0, stream>>>(
        seed, 0, dense, (long long)MAXSET * CDIM, counts, outv, CDIM, pm, B);
    mab_kernel<1, 1, false, false, false><<<B, 128, 0, stream>>>(
        outv, CDIM, nullptr, 0, nullptr, outv, CDIM, dc, B);
    maskzero_kernel<<<(B * CDIM + 255) / 256, 256, 0, stream>>>(outv, counts, B);
  };

  // 1) H_r = SetTransformer(gene_x[gpr_gene_idx] grouped by gpr_rxn_idx)
  run_st(gene_x, gpr_rxn, gpr_gene, E_GPR, N_RXN, ST_GR, H_r);

  // 2) Z_m = clamp-norm(emb); 2x stoichiometric conv
  clampnorm_kernel<<<N_MET, 32, 0, stream>>>(emb, Zm_a, N_MET);
  float* zin = Zm_a;
  float* zout = Zm_b;
  for (int l = 0; l < 2; ++l) {
    const bf16* lw = wb[CONV + l * 3 + 0];
    const float* av = F(CONV + l * 3 + 1);
    const float* bs = F(CONV + l * 3 + 2);
    gemm_rows_kernel<<<(N_MET + 63) / 64, 128, 0, stream>>>(zin, lw, xt, N_MET);
    gemm_rows_kernel<<<(N_RXN + 63) / 64, 128, 0, stream>>>(H_r, lw, et, N_RXN);
    hipMemsetAsync(segmax, 0, (size_t)N_RXN * 4, stream);
    hipMemsetAsync(segsum, 0, (size_t)N_RXN * 4, stream);
    hipMemsetAsync(Dn, 0, (size_t)N_MET * 4, stream);
    hipMemsetAsync(Be, 0, (size_t)N_RXN * 4, stream);
    hipMemsetAsync(e_out, 0, (size_t)N_RXN * CDIM * 4, stream);
    hipMemsetAsync(zout, 0, (size_t)N_MET * CDIM * 4, stream);
    edge_logits_kernel<<<(E_MET + 255) / 256, 256, 0, stream>>>(xt, et, av, met_met,
                                                                met_rxn, la, segmax, E_MET);
    expsum_kernel<<<(E_MET + 255) / 256, 256, 0, stream>>>(la, segmax, segsum, met_rxn, E_MET);
    deg_kernel<<<(E_MET + 255) / 256, 256, 0, stream>>>(stoich, met_met, met_rxn, Dn, Be, E_MET);
    eout_kernel<<<E_MET, 128, 0, stream>>>(xt, la, segsum, Be, stoich, met_met, met_rxn,
                                           e_out, E_MET);
    nodeout_kernel<<<E_MET, 128, 0, stream>>>(e_out, la, segsum, Dn, stoich, met_met,
                                              met_rxn, zout, E_MET);
    addbias_kernel<<<(N_MET * CDIM + 255) / 256, 256, 0, stream>>>(zout, bs, N_MET);
    float* t = zin; zin = zout; zout = t;
  }

  // 3) Z_r = SetTransformer(Z_m[met_met_idx] grouped by met_rxn_idx)
  run_st(zin, met_rxn, met_met, E_MET, N_RXN, ST_RM, Z_r);

  // 4) Z_mg = SetTransformer(Z_r[gpr_rxn_idx] grouped by gpr_gene_idx) -> output
  run_st(Z_r, gpr_gene, gpr_rxn, E_GPR, N_GENE, ST_GR, (float*)d_out);
}
